// HyperbolicGraphConstructor2_21964462752547
// MI455X (gfx1250) — compile-verified, matrix operand order checked by reading, not run on hardware
//
#include <hip/hip_runtime.h>
#include <hip/hip_bf16.h>
#include <math.h>

// ---------------- sizes ----------------
#define NN   2048
#define DD   10882
#define DPAD 10912          // 32*341, padded transpose target
#define NB2  341            // DPAD/32  : 32-wide N blocks
#define MB2  64             // 2048/32  : 32-wide M blocks
#define SQRT_N 45.254834f   // sqrt(2048)

typedef __attribute__((ext_vector_type(16))) _Float16 v16h;
typedef __attribute__((ext_vector_type(8)))  _Float16 v8h;
typedef __attribute__((ext_vector_type(8)))  float    v8f;

// ---------------- block reductions (256 threads) ----------------
__device__ __forceinline__ float block_sum(float v, float* smem) {
    int t = threadIdx.x;
    smem[t] = v; __syncthreads();
    for (int s = 128; s > 0; s >>= 1) {
        if (t < s) smem[t] += smem[t + s];
        __syncthreads();
    }
    float r = smem[0]; __syncthreads();
    return r;
}
__device__ __forceinline__ float block_max(float v, float* smem) {
    int t = threadIdx.x;
    smem[t] = v; __syncthreads();
    for (int s = 128; s > 0; s >>= 1) {
        if (t < s) smem[t] = fmaxf(smem[t], smem[t + s]);
        __syncthreads();
    }
    float r = smem[0]; __syncthreads();
    return r;
}

// ---------------- K0: per-row hyperbolic stats + hyp_x(f16) ----------------
// One block per row i. sqdist(hyp_x_i, hyp_avg_i) collapses to three dot
// products; also emits v[i] (normalized rank-1 logits factor) and hyp_x f16.
__global__ void prep_kernel(const float* __restrict__ x,
                            const float* __restrict__ avg,
                            _Float16* __restrict__ hx16,
                            float* __restrict__ v) {
    const int i = blockIdx.x;
    const int t = threadIdx.x;
    const float* xr = x   + (size_t)i * DD;
    const float* ar = avg + (size_t)i * DD;
    float sx2 = 0.f, sa2 = 0.f, sxa = 0.f;
    for (int j = t; j < DD; j += 256) {
        float xv = xr[j], av = ar[j];
        sx2 += xv * xv; sa2 += av * av; sxa += xv * av;
    }
    __shared__ float smem[256];
    sx2 = block_sum(sx2, smem);
    sa2 = block_sum(sa2, smem);
    sxa = block_sum(sxa, smem);

    __shared__ float s_fx;
    if (t == 0) {
        const float maxn = 1.f - 4e-3f;            // (1-BALL_EPS)/sqrt(c), c=1
        float nx = fmaxf(sqrtf(sx2), 1e-15f);
        float na = fmaxf(sqrtf(sa2), 1e-15f);
        float hn = tanhf(nx), ha = tanhf(na);
        float fx = hn / nx;  if (hn > maxn) fx *= maxn / hn;   // expmap0 + proj
        float fa = ha / na;  if (ha > maxn) fa *= maxn / ha;
        float X2 = fx * fx * sx2;                  // ||hyp_x||^2
        float Y2 = fa * fa * sa2;                  // ||hyp_avg||^2
        float XY = fx * fa * sxa;                  // <hyp_x, hyp_avg>
        // mobius_add(-p1, p2): x=-hyp_x, y=hyp_avg  (c=1)
        float Aa  = 1.f - 2.f * XY + Y2;
        float Bb  = 1.f - X2;
        float den = fmaxf(1.f - 2.f * XY + X2 * Y2, 1e-15f);
        float num2 = Aa * Aa * X2 - 2.f * Aa * Bb * XY + Bb * Bb * Y2;
        float dn = sqrtf(fmaxf(num2, 0.f)) / den;  // ||mobius_add||
        float r  = fminf(dn, 1.f - 1e-7f);         // artanh clip
        float dist = 2.f * atanhf(r);
        float d2 = dist * dist;                    // sqdist
        v[i] = -d2 / fmaxf(d2 * SQRT_N, 1e-12f);   // row-normalized broadcast
        s_fx = fx;
    }
    __syncthreads();
    float fx = s_fx;
    for (int j = t; j < DD; j += 256)
        hx16[(size_t)i * DD + j] = (_Float16)(fx * xr[j]);
}

// ---------------- K1: row sums of lin_w ----------------
__global__ void rowsum_w_kernel(const float* __restrict__ w,
                                float* __restrict__ srow) {
    const int j = blockIdx.x;
    float s = 0.f;
    for (int k = threadIdx.x; k < NN; k += 256) s += w[(size_t)j * NN + k];
    __shared__ float smem[256];
    s = block_sum(s, smem);
    if (threadIdx.x == 0) srow[j] = s;
}

// ---------------- K2: column softmax of rank-1 logits -> a (f16) ----------------
// logits[i,j] = v[i]*srow[j] + b[j]; softmax over i (dim=0), per column j.
__global__ void softmax_col_kernel(const float* __restrict__ v,
                                   const float* __restrict__ srow,
                                   const float* __restrict__ b,
                                   _Float16* __restrict__ a16) {
    const int j = blockIdx.x;
    const int t = threadIdx.x;
    __shared__ float logits[NN];
    __shared__ float smem[256];
    const float sj = srow[j], bj = b[j];
    float mx = -INFINITY;
    for (int i = t; i < NN; i += 256) {
        float l = fmaf(v[i], sj, bj);
        logits[i] = l;
        mx = fmaxf(mx, l);
    }
    mx = block_max(mx, smem);
    float sum = 0.f;
    for (int i = t; i < NN; i += 256) {
        float e = expf(logits[i] - mx);
        logits[i] = e;
        sum += e;
    }
    sum = block_sum(sum, smem);
    float inv = 1.f / sum;
    for (int i = t; i < NN; i += 256)
        a16[(size_t)i * NN + j] = (_Float16)(logits[i] * inv);
}

// ---------------- K3: x_norm[m] = ||a_m||_2 ----------------
__global__ void rownorm_a_kernel(const _Float16* __restrict__ a16,
                                 float* __restrict__ xnorm) {
    const int m = blockIdx.x;
    float s = 0.f;
    for (int k = threadIdx.x; k < NN; k += 256) {
        float av = (float)a16[(size_t)m * NN + k];
        s += av * av;
    }
    __shared__ float smem[256];
    s = block_sum(s, smem);
    if (threadIdx.x == 0) xnorm[m] = fmaxf(sqrtf(s), 1e-15f);
}

// ---------------- K4: transpose hyp_x [K=2048, D] -> BT [DPAD, 2048] (f16) ----------
__global__ void transpose_kernel(const _Float16* __restrict__ hx16,
                                 _Float16* __restrict__ bt) {
    __shared__ _Float16 tile[32][33];
    const int tx = threadIdx.x;          // 0..31
    const int ty = threadIdx.y;          // 0..7
    const int n0 = blockIdx.x * 32;      // D direction
    const int k0 = blockIdx.y * 32;      // K direction
    #pragma unroll
    for (int r = 0; r < 4; ++r) {
        int k = k0 + ty + r * 8;
        int n = n0 + tx;
        tile[ty + r * 8][tx] = (n < DD) ? hx16[(size_t)k * DD + n] : (_Float16)0.0f;
    }
    __syncthreads();
    #pragma unroll
    for (int r = 0; r < 4; ++r) {
        int n = n0 + ty + r * 8;         // BT row (n < DPAD always)
        int k = k0 + tx;                 // contiguous writes along K
        bt[(size_t)n * NN + k] = tile[tx][ty + r * 8];
    }
}

// ---------------- K5: WMMA GEMM  out = a16 [M,K] @ BT^T [K,N] ----------------
// One wave = 32x32 output block = 2x2 register-blocked 16x16 WMMA tiles.
// Per K-step: 2 A-frags + 2 B-frags (8 global_load_b128/lane total) feed
// 4 v_wmma_f32_16x16x32_f16 -> 2x arithmetic intensity vs 1 tile/wave.
// Fragment layout (16-bit A 16x32): lane L holds row M=L&15; K-halves
// [kbase..kbase+7] and [kbase+16..kbase+23], kbase=(L<16)?0:8 -> two b128
// loads per fragment. B mirrors with column n = L&15 (BT is row-major [n][k]).
__device__ __forceinline__ v16h load_frag(const _Float16* p) {
    v8h lo = *(const v8h*)(p);
    v8h hi = *(const v8h*)(p + 16);
    return __builtin_shufflevector(lo, hi, 0,1,2,3,4,5,6,7,8,9,10,11,12,13,14,15);
}

__global__ void gemm_wmma_kernel(const _Float16* __restrict__ A,
                                 const _Float16* __restrict__ BT,
                                 float* __restrict__ out) {
    const int wave = blockIdx.x * (blockDim.x >> 5) + (threadIdx.x >> 5);
    const int lane = threadIdx.x & 31;
    const int nblk = wave % NB2;
    const int mblk = wave / NB2;
    if (mblk >= MB2) return;
    const int l16   = lane & 15;
    const int hi    = lane >> 4;         // 0 or 1
    const int kbase = hi * 8;

    const _Float16* a0 = A  + (size_t)(mblk * 32 + l16) * NN + kbase;
    const _Float16* a1 = a0 + (size_t)16 * NN;
    const _Float16* b0 = BT + (size_t)(nblk * 32 + l16) * NN + kbase;
    const _Float16* b1 = b0 + (size_t)16 * NN;

    v8f acc00 = {}, acc01 = {}, acc10 = {}, acc11 = {};
    #pragma unroll 2
    for (int kb = 0; kb < NN; kb += 32) {
        // prefetch ~1KB ahead; locality 3 -> near (WGP) scope: pull into all
        // cache levels, hiding L2->WGP$ latency (operands are L2-resident).
        __builtin_prefetch((const void*)(a0 + kb + 512), 0, 3);
        __builtin_prefetch((const void*)(a1 + kb + 512), 0, 3);
        __builtin_prefetch((const void*)(b0 + kb + 512), 0, 3);
        __builtin_prefetch((const void*)(b1 + kb + 512), 0, 3);
        v16h av0 = load_frag(a0 + kb);
        v16h av1 = load_frag(a1 + kb);
        v16h bv0 = load_frag(b0 + kb);
        v16h bv1 = load_frag(b1 + kb);
        acc00 = __builtin_amdgcn_wmma_f32_16x16x32_f16(false, av0, false, bv0, (short)0, acc00, false, false);
        acc01 = __builtin_amdgcn_wmma_f32_16x16x32_f16(false, av0, false, bv1, (short)0, acc01, false, false);
        acc10 = __builtin_amdgcn_wmma_f32_16x16x32_f16(false, av1, false, bv0, (short)0, acc10, false, false);
        acc11 = __builtin_amdgcn_wmma_f32_16x16x32_f16(false, av1, false, bv1, (short)0, acc11, false, false);
    }
    // C/D layout: lane L -> column n = L&15; VGPR r -> row m = r + 8*(L>>4)
    const int n0 = nblk * 32 + l16;
    const int n1 = n0 + 16;
    const int m0 = mblk * 32 + hi * 8;   // rows for acc0x
    const int m1 = m0 + 16;              // rows for acc1x
    if (n0 < DD) {
        float* o0 = out + (size_t)m0 * DD + n0;
        float* o1 = out + (size_t)m1 * DD + n0;
        #pragma unroll
        for (int r = 0; r < 8; ++r) { o0[(size_t)r * DD] = acc00[r]; o1[(size_t)r * DD] = acc10[r]; }
    }
    if (n1 < DD) {
        float* o0 = out + (size_t)m0 * DD + n1;
        float* o1 = out + (size_t)m1 * DD + n1;
        #pragma unroll
        for (int r = 0; r < 8; ++r) { o0[(size_t)r * DD] = acc01[r]; o1[(size_t)r * DD] = acc11[r]; }
    }
}

// ---------------- K6: per-row mx_norm -> combined epilogue scale ----------------
// mobius_matvec tail + logmap0 collapse to out[m,:] *= scale[m].
__global__ void rowscale_kernel(const float* __restrict__ out,
                                const float* __restrict__ xnorm,
                                float* __restrict__ scale) {
    const int m = blockIdx.x;
    float s = 0.f;
    const float* r = out + (size_t)m * DD;
    for (int j = threadIdx.x; j < DD; j += 256) { float v = r[j]; s += v * v; }
    __shared__ float smem[256];
    s = block_sum(s, smem);
    if (threadIdx.x == 0) {
        float mxn = fmaxf(sqrtf(s), 1e-15f);                 // _norm(mx)
        float xn  = xnorm[m];                                // _norm(a_m)
        float g   = mxn / xn * atanhf(fminf(xn, 1.f - 1e-7f));
        float th  = tanhf(g);
        float s1  = th / mxn;                                // hyp_out = s1 * mx
        float nh  = fmaxf(th, 1e-15f);                       // ||hyp_out|| = tanh(g)
        float s2  = atanhf(fminf(nh, 1.f - 1e-7f)) / nh;     // logmap0 scale
        scale[m]  = s1 * s2;
    }
}

// ---------------- K7: in-place apply scale ----------------
__global__ void apply_scale_kernel(float* __restrict__ out,
                                   const float* __restrict__ scale) {
    const int m = blockIdx.y;
    const int j = blockIdx.x * 256 + threadIdx.x;
    if (j < DD) out[(size_t)m * DD + j] *= scale[m];
}

// ---------------- launcher ----------------
extern "C" void kernel_launch(void* const* d_in, const int* in_sizes, int n_in,
                              void* d_out, int out_size, void* d_ws, size_t ws_size,
                              hipStream_t stream) {
    // inputs: 0 idx (unused), 1 dist_metrix (unused), 2 x, 3 avg_metrix, 4 lin_w, 5 lin_b
    const float* x    = (const float*)d_in[2];
    const float* avg  = (const float*)d_in[3];
    const float* linw = (const float*)d_in[4];
    const float* linb = (const float*)d_in[5];
    float* out = (float*)d_out;

    char* ws = (char*)d_ws;
    size_t off = 0;
    _Float16* bt    = (_Float16*)(ws + off); off += (size_t)DPAD * NN * sizeof(_Float16); // 44.7 MB
    _Float16* hx16  = (_Float16*)(ws + off); off += (size_t)NN * DD  * sizeof(_Float16);  // 44.6 MB
    _Float16* a16   = (_Float16*)(ws + off); off += (size_t)NN * NN  * sizeof(_Float16);  //  8.4 MB
    float*    v     = (float*)(ws + off);    off += NN * sizeof(float);
    float*    srow  = (float*)(ws + off);    off += NN * sizeof(float);
    float*    xnorm = (float*)(ws + off);    off += NN * sizeof(float);
    float*    scale = (float*)(ws + off);    off += NN * sizeof(float);

    // 1) per-row hyperbolic stats, v[i], hyp_x -> f16
    prep_kernel<<<NN, 256, 0, stream>>>(x, avg, hx16, v);
    // 2) lin_w row sums
    rowsum_w_kernel<<<NN, 256, 0, stream>>>(linw, srow);
    // 3) column softmax -> a (f16, row-major [M,K])
    softmax_col_kernel<<<NN, 256, 0, stream>>>(v, srow, linb, a16);
    // 4) row norms of a
    rownorm_a_kernel<<<NN, 256, 0, stream>>>(a16, xnorm);
    // 5) transpose hyp_x -> BT [DPAD, K] (zero-padded)
    transpose_kernel<<<dim3(DPAD / 32, NN / 32), dim3(32, 8), 0, stream>>>(hx16, bt);
    // 6) WMMA GEMM: out = a @ hyp_x (2x2 register-blocked, 8 waves/block)
    {
        int total_waves = MB2 * NB2;             // 21824
        int blocks      = (total_waves + 7) / 8; // 2728
        gemm_wmma_kernel<<<blocks, 256, 0, stream>>>(a16, bt, out);
    }
    // 7) epilogue scale per row, 8) apply in place
    rowscale_kernel<<<NN, 256, 0, stream>>>(out, xnorm, scale);
    apply_scale_kernel<<<dim3((DD + 255) / 256, NN), 256, 0, stream>>>(out, scale);
}
